// ParameterGeneration_14714557956735
// MI455X (gfx1250) — matrix operand order repaired and checked by reading, not compiled, not run on hardware
//
#include <hip/hip_runtime.h>
#include <hip/hip_bf16.h>
#include <math.h>

typedef __attribute__((ext_vector_type(16))) __bf16 v16bf;
typedef __attribute__((ext_vector_type(8)))  __bf16 v8bf;
typedef __attribute__((ext_vector_type(8)))  float  v8f;

#define N_NODES 384
#define IN_F    128
#define HID     256
#define INT_OUT 800
#define HALF_OUT 400
#define NN400   (384*400)                 // row stride within K (or Q) block
#define KSIZE   ((size_t)384*384*400)     // elements in K block = 58,982,400

// ---------------------------------------------------------------------------
// Generic small GEMM: Y[M,N] = X[M,K] @ W^T + bias ; W rows are length-ldw
// (lets us slice fci_w1 into its two [256,256] halves with ldw=512).
// ---------------------------------------------------------------------------
__global__ void gemm_xwt_bias(const float* __restrict__ X,
                              const float* __restrict__ W,
                              const float* __restrict__ bias,
                              float* __restrict__ Y,
                              int M, int N, int K, int ldw) {
    int idx = blockIdx.x * blockDim.x + threadIdx.x;
    if (idx >= M * N) return;
    int m = idx / N, n = idx % N;
    const float* x = X + (size_t)m * K;
    const float* w = W + (size_t)n * ldw;
    float acc = bias ? bias[n] : 0.0f;
    for (int k = 0; k < K; k += 4) {
        acc += x[k]   * w[k]   + x[k+1] * w[k+1]
             + x[k+2] * w[k+2] + x[k+3] * w[k+3];
    }
    Y[idx] = acc;
}

// ---------------------------------------------------------------------------
// Attention: one block per (query n, head h). 128 threads.
// qkv row layout: [q(0..255) | k(256..511) | v(512..767)], head h = 32 lanes.
// ---------------------------------------------------------------------------
__global__ void __launch_bounds__(128)
attention_kernel(const float* __restrict__ qkv, float* __restrict__ out) {
    const int n = blockIdx.x;
    const int h = blockIdx.y;
    const int t = threadIdx.x;

    __shared__ float sc[N_NODES];
    __shared__ float red[128];
    __shared__ float qs[32];

    if (t < 32) qs[t] = qkv[(size_t)n * 768 + h * 32 + t];
    __syncthreads();

    const float scale = 0.1767766952966369f;  // 1/sqrt(32)
    for (int m = t; m < N_NODES; m += 128) {
        const float* kr = qkv + (size_t)m * 768 + 256 + h * 32;
        float acc = 0.0f;
        #pragma unroll
        for (int d = 0; d < 32; ++d) acc += qs[d] * kr[d];
        sc[m] = acc * scale;
    }
    __syncthreads();

    float lm = -INFINITY;
    for (int m = t; m < N_NODES; m += 128) lm = fmaxf(lm, sc[m]);
    red[t] = lm; __syncthreads();
    for (int s = 64; s > 0; s >>= 1) {
        if (t < s) red[t] = fmaxf(red[t], red[t + s]);
        __syncthreads();
    }
    const float mx = red[0];
    __syncthreads();

    float ls = 0.0f;
    for (int m = t; m < N_NODES; m += 128) {
        float e = __expf(sc[m] - mx);
        sc[m] = e;
        ls += e;
    }
    red[t] = ls; __syncthreads();
    for (int s = 64; s > 0; s >>= 1) {
        if (t < s) red[t] += red[t + s];
        __syncthreads();
    }
    const float inv = 1.0f / red[0];

    if (t < 32) {
        float acc = 0.0f;
        for (int m = 0; m < N_NODES; ++m)
            acc += sc[m] * qkv[(size_t)m * 768 + 512 + h * 32 + t];
        out[(size_t)n * HID + h * 32 + t] = acc * inv;
    }
}

// ---------------------------------------------------------------------------
// node = h2 @ fcn_w^T + fcn_b, scattered into W0/B0/W1 slots of d_out.
// ---------------------------------------------------------------------------
__global__ void node_kernel(const float* __restrict__ h2,
                            const float* __restrict__ fcn_w,
                            const float* __restrict__ fcn_b,
                            float* __restrict__ out) {
    int idx = blockIdx.x * blockDim.x + threadIdx.x;
    if (idx >= N_NODES * 80) return;
    int m = idx / 80, n = idx % 80;
    const float* x = h2 + (size_t)m * HID;
    const float* w = fcn_w + (size_t)n * HID;
    float acc = fcn_b[n];
    for (int k = 0; k < HID; k += 4)
        acc += x[k]*w[k] + x[k+1]*w[k+1] + x[k+2]*w[k+2] + x[k+3]*w[k+3];

    size_t base = 2 * KSIZE;
    if (n < 64)       out[base + (size_t)m * 64 + n] = acc;                        // W0
    else if (n < 72)  out[base + 384*64 + (size_t)m * 8 + (n - 64)] = acc;         // B0
    else              out[base + 384*64 + 384*8 + (size_t)m * 8 + (n - 72)] = acc; // W1
}

__global__ void f32_to_bf16(const float* __restrict__ src,
                            __bf16* __restrict__ dst, int n) {
    int i = blockIdx.x * blockDim.x + threadIdx.x;
    if (i < n) dst[i] = (__bf16)src[i];
}

// ---------------------------------------------------------------------------
// Dominant kernel: kq[p, :800] = tanh(a[j] + b[i] + b1) @ w2^T + b2,
// p = i*384 + j, bf16 WMMA with f32 accumulate.
//
// Block = 800 threads (25 waves); owns 64 pair-rows x ALL 800 columns.
// A-tile (64x256 bf16, 32KB LDS) built once; since 384 % 64 == 0, i is
// constant per block and j never wraps. Wave w owns N-tiles {2w, 2w+1}
// and M-subtiles {0..3}: per K-step it loads 4 A frags + 2 B frags, then
// issues 8 back-to-back WMMAs — each B fragment is reused by 4 WMMAs,
// cutting fci_w2 L2 traffic 4x vs. a 16-row block.
// Grid = (147456/64, 1). 64 v_wmma per wave.
// ---------------------------------------------------------------------------
__global__ void __launch_bounds__(800)
kq_wmma_kernel(const float* __restrict__ a,     // [384,256], indexed by j
               const float* __restrict__ bvec,  // [384,256], indexed by i
               const float* __restrict__ b1,    // [256]
               const __bf16* __restrict__ w2,   // [800,256] bf16 (B col-major)
               const float* __restrict__ b2,    // [800]
               float* __restrict__ out) {
    __shared__ __align__(16) __bf16 At[64][HID];   // 32 KB

    const int rowBase = blockIdx.x * 64;
    const int i0 = rowBase / N_NODES;      // constant for whole block
    const int j0 = rowBase - i0 * N_NODES; // multiple of 64; j0+63 < 384

    // Build fused-tanh A tile (64 pairs x 256) in bf16 — once per block.
    {
        const float* brow = bvec + (size_t)i0 * HID;
        for (int idx = threadIdx.x; idx < 64 * HID; idx += 800) {
            int r = idx >> 8;
            int k = idx & 255;
            float t = tanhf(a[(size_t)(j0 + r) * HID + k] + brow[k] + b1[k]);
            At[r][k] = (__bf16)t;
        }
    }
    __syncthreads();

    const int wave = threadIdx.x >> 5;   // 0..24
    const int lane = threadIdx.x & 31;
    const int half = lane >> 4;          // K-subset selector
    const int col  = lane & 15;          // A: row-in-subtile; B: column N

    // Two N-tiles per wave; 25*2 = 50 tiles, exact.
    const __bf16* wrow0 = w2 + (size_t)((wave * 2 + 0) * 16 + col) * HID;
    const __bf16* wrow1 = w2 + (size_t)((wave * 2 + 1) * 16 + col) * HID;

    v8f c[2][4];
    #pragma unroll
    for (int nt = 0; nt < 2; ++nt)
        #pragma unroll
        for (int mt = 0; mt < 4; ++mt) c[nt][mt] = (v8f){};

    #pragma unroll
    for (int ks = 0; ks < 8; ++ks) {
        const int k0 = ks * 32;

        // 4 A fragments (contiguous K order, same as B): 16 contiguous K halves per lane.
        v16bf af[4];
        #pragma unroll
        for (int mt = 0; mt < 4; ++mt) {
            const __bf16* arow = &At[mt * 16 + col][0];
            v8bf alo = *(const v8bf*)(arow + k0 + half * 16);
            v8bf ahi = *(const v8bf*)(arow + k0 + half * 16 + 8);
            af[mt] = __builtin_shufflevector(alo, ahi,
                         0,1,2,3,4,5,6,7,8,9,10,11,12,13,14,15);
        }

        // 2 B fragments (contiguous K order, same as A): 16 contiguous K halves per lane.
        v16bf bf[2];
        {
            v8bf l0 = *(const v8bf*)(wrow0 + k0 + half * 16);
            v8bf h0 = *(const v8bf*)(wrow0 + k0 + half * 16 + 8);
            bf[0] = __builtin_shufflevector(l0, h0,
                        0,1,2,3,4,5,6,7,8,9,10,11,12,13,14,15);
            v8bf l1 = *(const v8bf*)(wrow1 + k0 + half * 16);
            v8bf h1 = *(const v8bf*)(wrow1 + k0 + half * 16 + 8);
            bf[1] = __builtin_shufflevector(l1, h1,
                        0,1,2,3,4,5,6,7,8,9,10,11,12,13,14,15);
        }

        // 8 WMMAs; each B fragment reused 4x.
        #pragma unroll
        for (int nt = 0; nt < 2; ++nt)
            #pragma unroll
            for (int mt = 0; mt < 4; ++mt)
                c[nt][mt] = __builtin_amdgcn_wmma_f32_16x16x32_bf16(
                    false, af[mt], false, bf[nt], (short)0, c[nt][mt],
                    false, false);
    }

    // C layout: VGPR r -> M = half*8 + r (within subtile), N = col.
    // Add bias, route K vs Q halves, non-temporal stores (472 MB > L2).
    const size_t baseI = (size_t)i0 * NN400 + (size_t)j0 * HALF_OUT;
    #pragma unroll
    for (int nt = 0; nt < 2; ++nt) {
        const int ng = (wave * 2 + nt) * 16 + col;
        const float bias = b2[ng];
        const size_t cadd = (ng < HALF_OUT) ? (size_t)ng
                                            : KSIZE + (size_t)(ng - HALF_OUT);
        #pragma unroll
        for (int mt = 0; mt < 4; ++mt) {
            const size_t mbase = baseI + (size_t)(mt * 16 + half * 8) * HALF_OUT;
            #pragma unroll
            for (int r = 0; r < 8; ++r) {
                float val = c[nt][mt][r] + bias;
                __builtin_nontemporal_store(val,
                    out + mbase + (size_t)r * HALF_OUT + cadd);
            }
        }
    }
}

// ---------------------------------------------------------------------------
extern "C" void kernel_launch(void* const* d_in, const int* in_sizes, int n_in,
                              void* d_out, int out_size, void* d_ws, size_t ws_size,
                              hipStream_t stream) {
    const float* h_in   = (const float*)d_in[0];
    const float* fc_w   = (const float*)d_in[1];
    const float* fc_b   = (const float*)d_in[2];
    const float* inp_w  = (const float*)d_in[3];
    const float* inp_b  = (const float*)d_in[4];
    const float* outp_w = (const float*)d_in[5];
    const float* outp_b = (const float*)d_in[6];
    const float* fci_w1 = (const float*)d_in[7];
    const float* fci_b1 = (const float*)d_in[8];
    const float* fci_w2 = (const float*)d_in[9];
    const float* fci_b2 = (const float*)d_in[10];
    const float* fcn_w  = (const float*)d_in[11];
    const float* fcn_b  = (const float*)d_in[12];
    float* out = (float*)d_out;

    float* ws   = (float*)d_ws;
    float* h1   = ws;                  // 384*256
    float* qkv  = h1  + 384 * 256;     // 384*768
    float* ao   = qkv + 384 * 768;     // 384*256
    float* h2   = ao  + 384 * 256;     // 384*256
    float* av   = h2  + 384 * 256;     // 384*256  (h2 @ wa^T, indexed by j)
    float* bv   = av  + 384 * 256;     // 384*256  (h2 @ wb^T, indexed by i)
    __bf16* w2bf = (__bf16*)(bv + 384 * 256);   // 800*256 bf16

    const int TB = 256;

    // 1) h1 = h @ fc_w^T + fc_b            [384,128]x[128,256]
    gemm_xwt_bias<<<(384*256 + TB-1)/TB, TB, 0, stream>>>(
        h_in, fc_w, fc_b, h1, 384, 256, 128, 128);

    // 2) qkv = h1 @ in_proj_w^T + b        [384,256]x[256,768]
    gemm_xwt_bias<<<(384*768 + TB-1)/TB, TB, 0, stream>>>(
        h1, inp_w, inp_b, qkv, 384, 768, 256, 256);

    // 3) attention -> ao [384,256]
    attention_kernel<<<dim3(384, 8), 128, 0, stream>>>(qkv, ao);

    // 4) h2 = ao @ out_proj^T + b
    gemm_xwt_bias<<<(384*256 + TB-1)/TB, TB, 0, stream>>>(
        ao, outp_w, outp_b, h2, 384, 256, 256, 256);

    // 5) av = h2 @ wa^T ; bv = h2 @ wb^T   (fci_w1 is [256,512]; ldw=512)
    gemm_xwt_bias<<<(384*256 + TB-1)/TB, TB, 0, stream>>>(
        h2, fci_w1,        nullptr, av, 384, 256, 256, 512);
    gemm_xwt_bias<<<(384*256 + TB-1)/TB, TB, 0, stream>>>(
        h2, fci_w1 + 256,  nullptr, bv, 384, 256, 256, 512);

    // 6) node head -> W0/B0/W1 slots of d_out
    node_kernel<<<(384*80 + TB-1)/TB, TB, 0, stream>>>(h2, fcn_w, fcn_b, out);

    // 7) fci_w2 -> bf16
    f32_to_bf16<<<(INT_OUT*HID + TB-1)/TB, TB, 0, stream>>>(
        fci_w2, w2bf, INT_OUT * HID);

    // 8) dominant pairwise WMMA GEMM -> K and Q blocks of d_out
    kq_wmma_kernel<<<dim3(147456/64, 1), 800, 0, stream>>>(
        av, bv, fci_b1, w2bf, fci_b2, out);
}
